// Mixer2dTriU_49194555408700
// MI455X (gfx1250) — matrix operand order repaired and checked by reading, not compiled, not run on hardware
//
#include <hip/hip_runtime.h>
#include <hip/hip_bf16.h>

// Mixer2dTriU for MI455X (gfx1250): one workgroup per batch plane [T=128, C=128].
// bf16 WMMA (v_wmma_f32_16x16x32_bf16) for all three 128x128x128 matmuls,
// operands staged K-contiguous in LDS (135 KB of the 320 KB/WGP pool).

typedef __attribute__((ext_vector_type(16))) __bf16 v16bf;
typedef __attribute__((ext_vector_type(8)))  float  v8f;

constexpr int T = 128;
constexpr int C = 128;
constexpr int STRIDE = 136;             // bf16 elems per LDS row; 272B = 16B-aligned, skews banks
constexpr int NTHREADS = 256;           // 8 wave32 waves
constexpr size_t SMEM_BYTES =
    2u * 128u * STRIDE * 2u             // bufA, bufB (bf16 operands)
  + 128u * 128u * 4u                    // xf: x2 kept in f32 for final residual
  + 64u * 4u;                           // reduction scratch

union Frag { uint4 q[2]; v16bf v; };

__device__ __forceinline__ unsigned short f2bf(float f) {
  union { float f; unsigned int u; } v; v.f = f;
  unsigned int r = v.u + 0x7FFFu + ((v.u >> 16) & 1u);   // round-to-nearest-even
  return (unsigned short)(r >> 16);
}

// Block-wide mean/rstd over the 16384-element plane (wave shfl + LDS).
__device__ __forceinline__ void block_stats(float s, float s2, float* red, int tid,
                                            float& mu, float& rstd) {
  #pragma unroll
  for (int off = 16; off > 0; off >>= 1) {
    s  += __shfl_xor(s,  off, 32);
    s2 += __shfl_xor(s2, off, 32);
  }
  int lane = tid & 31, wid = tid >> 5;
  if (lane == 0) { red[wid] = s; red[8 + wid] = s2; }
  __syncthreads();
  if (tid == 0) {
    float S = 0.f, S2 = 0.f;
    #pragma unroll
    for (int i = 0; i < 8; ++i) { S += red[i]; S2 += red[8 + i]; }
    float m = S * (1.0f / 16384.0f);
    float var = S2 * (1.0f / 16384.0f) - m * m;
    red[16] = m;
    red[17] = rsqrtf(var + 1e-5f);
  }
  __syncthreads();
  mu = red[16]; rstd = red[17];
}

// 128x128x128 matmul: A[m][k] (LDS, bf16), Bt[n][k] (LDS, bf16), acc += A*B.
// Wave `wid` owns M-tile strip wid*16..wid*16+15; 8 N-tiles; 4 K-steps of 32.
__device__ __forceinline__ void mma_128(const unsigned short* A,
                                        const unsigned short* Bt,
                                        v8f acc[8], int wid, int lane) {
  const int mrow = wid * 16 + (lane & 15);
  const int half = lane >> 4;
  #pragma unroll
  for (int ks = 0; ks < 4; ++ks) {
    const int kb = ks * 32;
    Frag a;
    // 16-bit A layout: VGPR0-3 = K kb+half*8 .. +7 ; VGPR4-7 = K kb+16+half*8 .. +7
    a.q[0] = *(const uint4*)(A + mrow * STRIDE + kb + half * 8);
    a.q[1] = *(const uint4*)(A + mrow * STRIDE + kb + 16 + half * 8);
    #pragma unroll
    for (int nt = 0; nt < 8; ++nt) {
      const int nrow = nt * 16 + (lane & 15);
      Frag bfr;
      // B layout (matches A): VGPR0-3 = K kb+half*8 .. +7 ; VGPR4-7 = K kb+16+half*8 .. +7
      bfr.q[0] = *(const uint4*)(Bt + nrow * STRIDE + kb + half * 8);
      bfr.q[1] = *(const uint4*)(Bt + nrow * STRIDE + kb + 16 + half * 8);
      acc[nt] = __builtin_amdgcn_wmma_f32_16x16x32_bf16(
          false, a.v, false, bfr.v, (short)0, acc[nt], false, false);
    }
  }
}

__global__ __launch_bounds__(NTHREADS)
void mixer2d_triu_kernel(const float* __restrict__ x,
                         const float* __restrict__ g1, const float* __restrict__ b1ln,
                         const float* __restrict__ g2, const float* __restrict__ b2ln,
                         const float* __restrict__ Wt, const float* __restrict__ bt,
                         const float* __restrict__ W1, const float* __restrict__ b1,
                         const float* __restrict__ W2, const float* __restrict__ b2,
                         float* __restrict__ out) {
  extern __shared__ unsigned char smem_raw[];
  unsigned short* bufA = (unsigned short*)smem_raw;          // [128][STRIDE] bf16
  unsigned short* bufB = bufA + 128 * STRIDE;                // [128][STRIDE] bf16
  float* xf  = (float*)(bufB + 128 * STRIDE);                // [128][128] f32 (x2)
  float* red = xf + 128 * 128;

  const int bidx = blockIdx.x;
  const int tid  = threadIdx.x;
  const int lane = tid & 31;
  const int wid  = tid >> 5;
  const float* xb = x + (size_t)bidx * (T * C);

  // ---------- LayerNorm 1: stats over whole plane ----------
  float s = 0.f, s2 = 0.f;
  for (int i = 0; i < 64; ++i) {
    float v = xb[i * 256 + tid];
    s += v; s2 += v * v;
  }
  float mu, rstd;
  block_stats(s, s2, red, tid, mu, rstd);

  // h = (x-mu)*rstd*g1+b1 -> bufB transposed [c][t] (B operand, K=j contiguous)
  for (int i = 0; i < 64; ++i) {
    int idx = i * 256 + tid;
    int t = idx >> 7, c = idx & 127;
    float h = (xb[idx] - mu) * rstd * g1[idx] + b1ln[idx];
    bufB[c * STRIDE + t] = f2bf(h);
  }
  // Wm = tril(Wt) -> bufA [t][j], paired bf16 stores
  for (int p = 0; p < 32; ++p) {
    int pidx = (p * 256 + tid) * 2;
    int t = pidx >> 7, j = pidx & 127;
    float w0 = (j     <= t) ? Wt[pidx]     : 0.f;
    float w1 = (j + 1 <= t) ? Wt[pidx + 1] : 0.f;
    unsigned int pk = (unsigned int)f2bf(w0) | ((unsigned int)f2bf(w1) << 16);
    *(unsigned int*)(bufA + t * STRIDE + j) = pk;
  }
  __syncthreads();

  // ---------- matmul 1: tm = Wm @ h ----------
  v8f acc[8];
  const v8f vzero = {0.f, 0.f, 0.f, 0.f, 0.f, 0.f, 0.f, 0.f};
  #pragma unroll
  for (int nt = 0; nt < 8; ++nt) acc[nt] = vzero;
  mma_128(bufA, bufB, acc, wid, lane);

  // ---------- residual r = tm + bt + x ; LayerNorm 2 stats ----------
  const int mbase = wid * 16 + 8 * (lane >> 4);   // C/D: VGPR v -> row mbase+v
  float btv[8];
  #pragma unroll
  for (int v = 0; v < 8; ++v) btv[v] = bt[mbase + v];
  s = 0.f; s2 = 0.f;
  #pragma unroll
  for (int nt = 0; nt < 8; ++nt) {
    int c = nt * 16 + (lane & 15);
    #pragma unroll
    for (int v = 0; v < 8; ++v) {
      int t = mbase + v;
      float r = acc[nt][v] + btv[v] + xb[t * 128 + c];
      acc[nt][v] = r;
      s += r; s2 += r * r;
    }
  }
  float mu2, rstd2;
  block_stats(s, s2, red, tid, mu2, rstd2);
  // (all bufA/bufB reads of matmul1 completed before the barrier inside block_stats)

  // ---------- x2 = norm(r)*g2+b2 : f32 -> xf, bf16 -> bufA ; W1 -> bufB ----------
  #pragma unroll
  for (int nt = 0; nt < 8; ++nt) {
    int c = nt * 16 + (lane & 15);
    #pragma unroll
    for (int v = 0; v < 8; ++v) {
      int t = mbase + v;
      int idx = t * 128 + c;
      float x2 = (acc[nt][v] - mu2) * rstd2 * g2[idx] + b2ln[idx];
      xf[idx] = x2;
      bufA[t * STRIDE + c] = f2bf(x2);
    }
  }
  for (int p = 0; p < 32; ++p) {
    int pidx = (p * 256 + tid) * 2;
    int o = pidx >> 7, c = pidx & 127;
    unsigned int pk = (unsigned int)f2bf(W1[pidx]) | ((unsigned int)f2bf(W1[pidx + 1]) << 16);
    *(unsigned int*)(bufB + o * STRIDE + c) = pk;   // W1[o][c]: already Bt layout for x2@W1^T
  }
  __syncthreads();

  // ---------- matmul 2: u = x2 @ W1^T ----------
  #pragma unroll
  for (int nt = 0; nt < 8; ++nt) acc[nt] = vzero;
  mma_128(bufA, bufB, acc, wid, lane);

  // ---------- exact GELU(u + b1) ----------
  #pragma unroll
  for (int nt = 0; nt < 8; ++nt) {
    int o = nt * 16 + (lane & 15);
    float bias = b1[o];
    #pragma unroll
    for (int v = 0; v < 8; ++v) {
      float u = acc[nt][v] + bias;
      acc[nt][v] = 0.5f * u * (1.0f + erff(u * 0.70710678118654752f));
    }
  }
  __syncthreads();   // done reading bufA/bufB for matmul 2

  // g -> bufA [t][o] ; W2 -> bufB
  #pragma unroll
  for (int nt = 0; nt < 8; ++nt) {
    int o = nt * 16 + (lane & 15);
    #pragma unroll
    for (int v = 0; v < 8; ++v)
      bufA[(mbase + v) * STRIDE + o] = f2bf(acc[nt][v]);
  }
  for (int p = 0; p < 32; ++p) {
    int pidx = (p * 256 + tid) * 2;
    int o = pidx >> 7, c = pidx & 127;
    unsigned int pk = (unsigned int)f2bf(W2[pidx]) | ((unsigned int)f2bf(W2[pidx + 1]) << 16);
    *(unsigned int*)(bufB + o * STRIDE + c) = pk;
  }
  __syncthreads();

  // ---------- matmul 3: y = g @ W2^T ; out = x2 + y + b2 ----------
  #pragma unroll
  for (int nt = 0; nt < 8; ++nt) acc[nt] = vzero;
  mma_128(bufA, bufB, acc, wid, lane);

  float* outb = out + (size_t)bidx * (T * C);
  #pragma unroll
  for (int nt = 0; nt < 8; ++nt) {
    int o = nt * 16 + (lane & 15);
    float bias2 = b2[o];
    #pragma unroll
    for (int v = 0; v < 8; ++v) {
      int t = mbase + v;
      outb[t * 128 + o] = xf[t * 128 + o] + acc[nt][v] + bias2;
    }
  }
}

extern "C" void kernel_launch(void* const* d_in, const int* in_sizes, int n_in,
                              void* d_out, int out_size, void* d_ws, size_t ws_size,
                              hipStream_t stream) {
  (void)in_sizes; (void)n_in; (void)d_ws; (void)ws_size; (void)out_size;
  const float* x   = (const float*)d_in[0];
  const float* g1  = (const float*)d_in[1];
  const float* b1l = (const float*)d_in[2];
  const float* g2  = (const float*)d_in[3];
  const float* b2l = (const float*)d_in[4];
  const float* Wt  = (const float*)d_in[5];
  const float* bt  = (const float*)d_in[6];
  const float* W1  = (const float*)d_in[7];
  const float* b1  = (const float*)d_in[8];
  const float* W2  = (const float*)d_in[9];
  const float* b2  = (const float*)d_in[10];
  float* out = (float*)d_out;

  // 135 KB dynamic LDS per workgroup (CDNA5: up to 320 KB/WG)
  (void)hipFuncSetAttribute((const void*)mixer2d_triu_kernel,
                            hipFuncAttributeMaxDynamicSharedMemorySize,
                            (int)SMEM_BYTES);
  mixer2d_triu_kernel<<<1024, NTHREADS, SMEM_BYTES, stream>>>(
      x, g1, b1l, g2, b2l, Wt, bt, W1, b1, W2, b2, out);
}
